// GraphAttentionLayer_37323265802321
// MI455X (gfx1250) — matrix operand order repaired and checked by reading, not compiled, not run on hardware
//
#include <hip/hip_runtime.h>
#include <hip/hip_bf16.h>
#include <math.h>

typedef __attribute__((ext_vector_type(16))) _Float16 v16h;
typedef __attribute__((ext_vector_type(8)))  float    v8f;

#define NN      8192
#define IN_F    512
#define OUT_F   64
#define NEG_INF (-9000000000000000.0f)
#define LRELU_A 0.2f

// ---------------------------------------------------------------------------
// Workspace layout (bytes):
//   Wt   f16 [64][512]     @ 0        (65536)
//   h32  f32 [8192][64]    @ 65536    (2097152)
//   hT   f16 [64][8192]    @ 2162688  (1048576)   (32B-aligned rows for v16h)
//   attL f32 [8192]        @ 3211264  (32768)
//   attR f32 [8192]        @ 3244032  (32768)
// ---------------------------------------------------------------------------
#define WS_WT   0
#define WS_H32  65536
#define WS_HT   2162688
#define WS_ATTL 3211264
#define WS_ATTR 3244032

// --- Kernel 0: W (512x64 f32, row-major) -> Wt (64x512 f16) ----------------
__global__ void prep_wt(const float* __restrict__ W, _Float16* __restrict__ Wt) {
    int idx = blockIdx.x * blockDim.x + threadIdx.x;     // 0 .. 512*64-1
    if (idx >= IN_F * OUT_F) return;
    int k = idx / OUT_F;
    int c = idx % OUT_F;
    Wt[(size_t)c * IN_F + k] = (_Float16)W[idx];
}

// --- Kernel 1: h = input @ W via v_wmma_f32_16x16x32_f16 --------------------
// one wave (32 thr) per 16-row tile; 4 accumulators cover N=64
__global__ __launch_bounds__(32)
void gemm_h(const float* __restrict__ input, const _Float16* __restrict__ Wt,
            float* __restrict__ h32, _Float16* __restrict__ hT) {
    const int rowbase = blockIdx.x * 16;
    const int lane    = threadIdx.x;        // 0..31
    const int lane16  = lane & 15;
    const int sel     = (lane >> 4) & 1;    // which lane-half

    v8f acc0 = {}, acc1 = {}, acc2 = {}, acc3 = {};

    const float* arow = input + (size_t)(rowbase + lane16) * IN_F;

    for (int kb = 0; kb < IN_F; kb += 32) {
        // A: 16x32 f16, lane = row; halves 0..7 -> K = kb + sel*16 + i,
        //                           halves 8..15 -> K = kb + sel*16 + 8 + i
        v16h av;
        const float* a0 = arow + kb + sel * 16;
        const float* a1 = a0 + 8;
#pragma unroll
        for (int i = 0; i < 8; ++i) {
            av[i]     = (_Float16)a0[i];
            av[8 + i] = (_Float16)a1[i];
        }
        // B: 32x16 f16 per N-tile; lane = col, K contiguous per lane-half
#pragma unroll
        for (int t = 0; t < 4; ++t) {
            v16h bv = *(const v16h*)(Wt + (size_t)(t * 16 + lane16) * IN_F + kb + sel * 16);
            v8f* cp = (t == 0) ? &acc0 : (t == 1) ? &acc1 : (t == 2) ? &acc2 : &acc3;
            *cp = __builtin_amdgcn_wmma_f32_16x16x32_f16(false, av, false, bv,
                                                         (short)0, *cp, false, false);
        }
    }

    // C layout: element r -> row = rowbase + r + 8*sel, col = t*16 + lane16
#pragma unroll
    for (int t = 0; t < 4; ++t) {
        v8f acc = (t == 0) ? acc0 : (t == 1) ? acc1 : (t == 2) ? acc2 : acc3;
        int col = t * 16 + lane16;
#pragma unroll
        for (int r = 0; r < 8; ++r) {
            int   row = rowbase + r + 8 * sel;
            float v   = acc[r];
            h32[(size_t)row * OUT_F + col]  = v;
            hT[(size_t)col * NN + row]      = (_Float16)v;   // transposed f16 copy
        }
    }
}

// --- Kernel 2: att_l/att_r from fp32 h (reference-precision logits) ---------
__global__ void att_kernel(const float* __restrict__ h32, const float* __restrict__ a,
                           float* __restrict__ attL, float* __restrict__ attR) {
    int i = blockIdx.x * blockDim.x + threadIdx.x;
    if (i >= NN) return;
    const float* hr = h32 + (size_t)i * OUT_F;
    float sl = 0.f, sr = 0.f;
#pragma unroll
    for (int c = 0; c < OUT_F; ++c) {
        float hv = hr[c];
        sl += hv * a[c];
        sr += hv * a[OUT_F + c];
    }
    attL[i] = sl;
    attR[i] = sr;
}

// --- Kernel 3: fused mask + online softmax + (P @ h) + ELU ------------------
// 512 blocks x 128 thr (4 waves). Block = 16 output rows; wave w = cols [16w,16w+16)
// of the output AND owns softmax state for rows {w, w+4, w+8, w+12}.
// Streams j in tiles of 64 (2 WMMA K-steps); adj read exactly once from HBM,
// next tile's adj loads issued before the barrier to overlap HBM with WMMA.
__global__ __launch_bounds__(128)
void attn_av(const int* __restrict__ adj, const _Float16* __restrict__ hT,
             const float* __restrict__ attL, const float* __restrict__ attR,
             float* __restrict__ out) {
    __shared__ __align__(32) _Float16 p_lds[16 * 64];   // p packed f16 (A operand)
    __shared__ float scale_lds[16];
    __shared__ float s_lds[16];

    const int tid     = threadIdx.x;        // 0..127
    const int wave    = tid >> 5;           // 0..3
    const int lane    = tid & 31;
    const int lane16  = lane & 15;
    const int sel     = (lane >> 4) & 1;
    const int rowbase = blockIdx.x * 16;

    // wave w owns rows w+4q (q=0..3); lane = j within 32-col half
    float aL[4];
#pragma unroll
    for (int q = 0; q < 4; ++q) aL[q] = attL[rowbase + wave + 4 * q];

    const int* adjrow[4];
#pragma unroll
    for (int q = 0; q < 4; ++q)
        adjrow[q] = adj + (size_t)(rowbase + wave + 4 * q) * NN + lane;

    float m[4], s[4];
#pragma unroll
    for (int q = 0; q < 4; ++q) { m[q] = NEG_INF; s[q] = 0.f; }

    v8f acc0 = {}, acc1 = {};   // two 16-col... (wave's 16-col slice, rows 0..15)
    const _Float16* hTn = hT + (size_t)(wave * 16 + lane16) * NN;

    // prefetch adj tile 0: 8 values/thread (4 rows x 2 column-halves)
    int adjv[4][2];
#pragma unroll
    for (int q = 0; q < 4; ++q) {
        adjv[q][0] = adjrow[q][0];
        adjv[q][1] = adjrow[q][32];
    }

    for (int jbase = 0; jbase < NN; jbase += 64) {
        float aR0 = attR[jbase + lane];
        float aR1 = attR[jbase + 32 + lane];

        // ---- logits + online-softmax update, all in registers/shuffles ----
#pragma unroll
        for (int q = 0; q < 4; ++q) {
            float x0 = aL[q] + aR0;  x0 = (x0 > 0.f) ? x0 : LRELU_A * x0;
            float x1 = aL[q] + aR1;  x1 = (x1 > 0.f) ? x1 : LRELU_A * x1;
            float e0 = (adjv[q][0] == 0) ? NEG_INF : x0;
            float e1 = (adjv[q][1] == 0) ? NEG_INF : x1;

            // wave-wide row max (butterfly, wave32)
            float rmax = fmaxf(e0, e1);
#pragma unroll
            for (int off = 16; off > 0; off >>= 1)
                rmax = fmaxf(rmax, __shfl_xor(rmax, off, 32));

            float m_new = fmaxf(m[q], rmax);
            float sc    = __expf(m[q] - m_new);
            m[q] = m_new;

            float p0 = __expf(e0 - m_new);
            float p1 = __expf(e1 - m_new);

            float psum = p0 + p1;
#pragma unroll
            for (int off = 16; off > 0; off >>= 1)
                psum += __shfl_xor(psum, off, 32);
            s[q] = s[q] * sc + psum;

            int row = wave + 4 * q;
            p_lds[row * 64 + lane]      = (_Float16)p0;
            p_lds[row * 64 + 32 + lane] = (_Float16)p1;
            if (lane == q) scale_lds[row] = sc;
        }

        // issue next tile's adj loads before the barrier (overlap with WMMA)
        if (jbase + 64 < NN) {
#pragma unroll
            for (int q = 0; q < 4; ++q) {
                adjv[q][0] = adjrow[q][jbase + 64];
                adjv[q][1] = adjrow[q][jbase + 96];
            }
        }
        __syncthreads();

        // ---- rescale accumulators, two K=32 WMMAs ----
        {
            float sc0[8];
#pragma unroll
            for (int r = 0; r < 8; ++r) sc0[r] = scale_lds[r + 8 * sel];
#pragma unroll
            for (int r = 0; r < 8; ++r) { acc0[r] *= sc0[r]; acc1[r] *= sc0[r]; }
            // acc0 and acc1 are K-partials of the SAME tile; sum at the end.
        }

#pragma unroll
        for (int kk = 0; kk < 2; ++kk) {
            // A: lane=row; halves 0..7 -> K=kk*32+sel*16+i, 8..15 -> +8
            const _Float16* pr = &p_lds[lane16 * 64 + kk * 32 + sel * 16];
            v16h av;
#pragma unroll
            for (int i = 0; i < 8; ++i) {
                av[i]     = pr[i];
                av[8 + i] = pr[8 + i];
            }
            v16h bv = *(const v16h*)(hTn + jbase + kk * 32 + sel * 16);
            if (kk == 0)
                acc0 = __builtin_amdgcn_wmma_f32_16x16x32_f16(false, av, false, bv,
                                                              (short)0, acc0, false, false);
            else
                acc1 = __builtin_amdgcn_wmma_f32_16x16x32_f16(false, av, false, bv,
                                                              (short)0, acc1, false, false);
        }
        __syncthreads();
    }

#pragma unroll
    for (int q = 0; q < 4; ++q)
        if (lane == q) s_lds[wave + 4 * q] = s[q];
    __syncthreads();

    // ---- epilogue: combine K-partials, divide by denom, ELU, write fp32 ----
    const int col = wave * 16 + lane16;
#pragma unroll
    for (int r = 0; r < 8; ++r) {
        int   row = rowbase + r + 8 * sel;
        float v   = (acc0[r] + acc1[r]) / s_lds[r + 8 * sel];
        v = (v > 0.f) ? v : (__expf(v) - 1.0f);
        out[(size_t)row * OUT_F + col] = v;
    }
}

// ---------------------------------------------------------------------------
extern "C" void kernel_launch(void* const* d_in, const int* in_sizes, int n_in,
                              void* d_out, int out_size, void* d_ws, size_t ws_size,
                              hipStream_t stream) {
    const float* input = (const float*)d_in[0];   // 8192 x 512 f32
    const int*   adj   = (const int*)d_in[1];     // 8192 x 8192 i32
    const float* W     = (const float*)d_in[2];   // 512 x 64 f32
    const float* a     = (const float*)d_in[3];   // 128 f32
    float*       out   = (float*)d_out;           // 8192 x 64 f32

    char* ws = (char*)d_ws;
    _Float16* Wt   = (_Float16*)(ws + WS_WT);
    float*    h32  = (float*)   (ws + WS_H32);
    _Float16* hT   = (_Float16*)(ws + WS_HT);
    float*    attL = (float*)   (ws + WS_ATTL);
    float*    attR = (float*)   (ws + WS_ATTR);

    prep_wt<<<(IN_F * OUT_F + 255) / 256, 256, 0, stream>>>(W, Wt);
    gemm_h<<<NN / 16, 32, 0, stream>>>(input, Wt, h32, hT);
    att_kernel<<<NN / 256, 256, 0, stream>>>(h32, a, attL, attR);
    attn_av<<<NN / 16, 128, 0, stream>>>(adj, hT, attL, attR, out);
}
